// GRU_8641474200231
// MI455X (gfx1250) — matrix operand order repaired and checked by reading, not compiled, not run on hardware
//
#include <hip/hip_runtime.h>
#include <hip/hip_bf16.h>

typedef __bf16 bf16_t;
typedef __attribute__((ext_vector_type(16))) __bf16 v16bf;
typedef __attribute__((ext_vector_type(8)))  __bf16 v8bf;
typedef __attribute__((ext_vector_type(8)))  float   v8f;
typedef __attribute__((ext_vector_type(4)))  int     v4i;

// Problem dims
#define B_   64
#define T_   512
#define D_   128
#define H_   512
#define G3H  1536   // 3*H

// Scan-kernel LDS strides (padded to dodge bank conflicts, keep 16B alignment)
#define GH_LD 1540            // f32 row stride for gates scratch
#define HF_LD 512             // f32 h-state row stride
#define HB_LD 528             // bf16 h-state row stride (1056B, 16B aligned)
#define SCAN_LDS_BYTES (16*GH_LD*4 + 16*HF_LD*4 + 16*HB_LD*2)  // 148224

// Input-GEMM tiling
#define KSLAB 128             // K elements staged per slab
#define BS_LD 136             // bf16 LDS row stride for staged B (272B, 16B aligned)

// CDNA5 async global->LDS path (guarded; sync fallback keeps compile safe)
#if defined(__HIP_DEVICE_COMPILE__) && __has_builtin(__builtin_amdgcn_global_load_async_to_lds_b128)
#define HAVE_ASYNC 1
#else
#define HAVE_ASYNC 0
#endif

#if HAVE_ASYNC
#if __has_builtin(__builtin_amdgcn_s_wait_asynccnt)
#define WAIT_ASYNC(n) __builtin_amdgcn_s_wait_asynccnt(n)
#else
#define WAIT_ASYNC(n)                                              \
    do {                                                           \
        if ((n) == 0) asm volatile("s_wait_asynccnt 0x0" ::: "memory"); \
        else          asm volatile("s_wait_asynccnt 0x4" ::: "memory"); \
    } while (0)
#endif
typedef __attribute__((address_space(1))) v4i* gptr_b128;
typedef __attribute__((address_space(3))) v4i* lptr_b128;
#endif

// ---------------------------------------------------------------------------
// fp32 -> bf16 conversion (grid-stride)
// ---------------------------------------------------------------------------
__global__ __launch_bounds__(256) void cvt_f32_bf16(const float* __restrict__ in,
                                                    bf16_t* __restrict__ out, int n) {
    int i = blockIdx.x * blockDim.x + threadIdx.x;
    int stride = gridDim.x * blockDim.x;
    for (; i < n; i += stride) out[i] = (bf16_t)in[i];
}

// ---------------------------------------------------------------------------
// WMMA fragment loaders (contiguous k = 16*(lane/16) + s in A and B, wave32)
// ---------------------------------------------------------------------------
// A: 16x32 bf16, row-major source, lda = K stride in elements.
__device__ __forceinline__ v16bf load_a_rm(const bf16_t* __restrict__ base, int lda,
                                           int m0, int k0, int lane) {
    const int m  = m0 + (lane & 15);
    const int hi = lane >> 4;
    const bf16_t* p = base + (size_t)m * lda + k0 + hi * 16;
    v8bf lo = *(const v8bf*)p;          // K offsets hi*16 + 0..7
    v8bf hh = *(const v8bf*)(p + 8);    // K offsets hi*16 + 8..15
    v16bf a;
#pragma unroll
    for (int i = 0; i < 8; ++i) { a[i] = lo[i]; a[8 + i] = hh[i]; }
    return a;
}

// B: 32x16 bf16 where B[k][n] = W[n][k], W row-major with ldw = K stride.
__device__ __forceinline__ v16bf load_b_w(const bf16_t* __restrict__ w, int ldw,
                                          int n0, int k0, int lane) {
    const int n  = n0 + (lane & 15);
    const int hi = lane >> 4;
    const bf16_t* p = w + (size_t)n * ldw + k0 + hi * 16;   // 16 contiguous K
    v8bf lo = *(const v8bf*)p;
    v8bf hh = *(const v8bf*)(p + 8);
    v16bf b;
#pragma unroll
    for (int i = 0; i < 8; ++i) { b[i] = lo[i]; b[8 + i] = hh[i]; }
    return b;
}

// Same B fragment but from an LDS-staged [64 x BS_LD] tile.
__device__ __forceinline__ v16bf load_b_lds(const bf16_t* __restrict__ bs,
                                            int nloc0, int k0, int lane) {
    const int r  = nloc0 + (lane & 15);
    const int hi = lane >> 4;
    const bf16_t* p = bs + r * BS_LD + k0 + hi * 16;
    v8bf lo = *(const v8bf*)p;
    v8bf hh = *(const v8bf*)(p + 8);
    v16bf b;
#pragma unroll
    for (int i = 0; i < 8; ++i) { b[i] = lo[i]; b[8 + i] = hh[i]; }
    return b;
}

// ---------------------------------------------------------------------------
// Input GEMM: out[M][N] = A[M][K](bf16) * W[N][K]^T(bf16) + bias[N]  (f32 out)
// Block = 128 rows x 64 cols. 8 waves, one 16x64 strip each, sharing a B strip
// staged in LDS via async global->LDS copies, double-buffered over 128-K slabs.
// ---------------------------------------------------------------------------
__global__ __launch_bounds__(256, 1) void wmma_gemm_bias(const bf16_t* __restrict__ A,
                                                         const bf16_t* __restrict__ W,
                                                         const float* __restrict__ bias,
                                                         float* __restrict__ out,
                                                         int M, int N, int K) {
    __shared__ bf16_t Bs[2][64 * BS_LD];   // 2 x 17408B

    const int lane = threadIdx.x & 31;
    const int wave = threadIdx.x >> 5;
    const int nStrips = N / 64;
    const int nstrip = blockIdx.x % nStrips;
    const int mblk   = blockIdx.x / nStrips;
    const int n0 = nstrip * 64;
    const int m0 = mblk * 128 + wave * 16;
    const int nslab = K / KSLAB;

    // Stage one K-slab (64 cols x 128 K) into LDS: 1024 x 16B chunks, 4/thread.
    auto stage = [&](int s, int buf) {
        const int ks = s * KSLAB;
#pragma unroll
        for (int q = 0; q < 4; ++q) {
            const int c  = threadIdx.x + q * 256;
            const int r  = c >> 4;             // 0..63 (column of W / row of strip)
            const int kc = (c & 15) * 8;       // 0..120 bf16 within slab
            const bf16_t* gp = W + (size_t)(n0 + r) * K + ks + kc;
            bf16_t* lp = &Bs[buf][r * BS_LD + kc];
#if HAVE_ASYNC
            __builtin_amdgcn_global_load_async_to_lds_b128(
                (gptr_b128)gp, (lptr_b128)lp, 0, 0);
#else
            *(v8bf*)lp = *(const v8bf*)gp;
#endif
        }
    };

    v8f acc[4];
#pragma unroll
    for (int j = 0; j < 4; ++j) acc[j] = (v8f){0.f,0.f,0.f,0.f,0.f,0.f,0.f,0.f};

    // One slab of compute: 4 K-steps of 32, B register-double-buffered, A
    // fragments issued up-front -> WMMAs overlap ds/global latency.
    auto compute = [&](int s) {
        const bf16_t* bs = &Bs[s & 1][0];
        const int ks = s * KSLAB;
        v16bf a[4];
#pragma unroll
        for (int q = 0; q < 4; ++q) a[q] = load_a_rm(A, K, m0, ks + q * 32, lane);
        v16bf bc[4];
#pragma unroll
        for (int j = 0; j < 4; ++j) bc[j] = load_b_lds(bs, j * 16, 0, lane);
#pragma unroll
        for (int q = 0; q < 4; ++q) {
            v16bf bn[4];
            if (q < 3) {
#pragma unroll
                for (int j = 0; j < 4; ++j)
                    bn[j] = load_b_lds(bs, j * 16, (q + 1) * 32, lane);
            }
#pragma unroll
            for (int j = 0; j < 4; ++j)
                acc[j] = __builtin_amdgcn_wmma_f32_16x16x32_bf16(
                    false, a[q], false, bc[j], (short)0, acc[j], false, false);
            if (q < 3) {
#pragma unroll
                for (int j = 0; j < 4; ++j) bc[j] = bn[j];
            }
        }
    };

    // Software pipeline, peeled epilogue: steady state always stages s+1.
    stage(0, 0);
    for (int s = 0; s + 1 < nslab; ++s) {
        stage(s + 1, (s + 1) & 1);
#if HAVE_ASYNC
        WAIT_ASYNC(4);      // slab s complete; slab s+1 (4 insts/thread) in flight
#endif
        __syncthreads();
        compute(s);
        __syncthreads();    // protect Bs[s&1] before it is re-staged
    }
#if HAVE_ASYNC
    WAIT_ASYNC(0);
#endif
    __syncthreads();
    compute(nslab - 1);

    const int hi = lane >> 4, nloc = lane & 15;
#pragma unroll
    for (int j = 0; j < 4; ++j) {
        const int n = n0 + j * 16 + nloc;
        const float bv = bias[n];
#pragma unroll
        for (int v = 0; v < 8; ++v) {
            const int m = m0 + v + 8 * hi;
            out[(size_t)m * N + n] = acc[j][v] + bv;
        }
    }
}

// ---------------------------------------------------------------------------
// Fused GRU recurrence for a 16-row batch slice. One WG = 16 batch rows,
// full H recursion in LDS, no inter-WG sync. blockDim = 256 (8 waves).
// ---------------------------------------------------------------------------
__device__ __forceinline__ float sig_f(float x) { return 1.f / (1.f + __expf(-x)); }
__device__ __forceinline__ float tanh_f(float x) {
    x = fminf(fmaxf(x, -15.f), 15.f);
    float e = __expf(2.f * x);
    return (e - 1.f) / (e + 1.f);
}

__global__ __launch_bounds__(256, 1) void gru_scan(const float* __restrict__ gx,   // [B*T][3H]
                                                   const bf16_t* __restrict__ Whh, // [3H][H]
                                                   const float* __restrict__ bhh,  // [3H]
                                                   bf16_t* __restrict__ hout) {    // [B*T][H]
    extern __shared__ char smem[];
    float*  gh = (float*)smem;                                      // [16][GH_LD]
    float*  hF = (float*)(smem + 16 * GH_LD * 4);                   // [16][HF_LD]
    bf16_t* hB = (bf16_t*)(smem + 16 * GH_LD * 4 + 16 * HF_LD * 4); // [16][HB_LD]

    const int lane = threadIdx.x & 31;
    const int wave = threadIdx.x >> 5;
    const int b0   = blockIdx.x * 16;

    for (int u = threadIdx.x; u < 16 * H_; u += blockDim.x) {
        const int m = u >> 9, j = u & (H_ - 1);
        hF[m * HF_LD + j] = 0.f;
        hB[m * HB_LD + j] = (bf16_t)0.f;
    }
    __syncthreads();

    const int mA  = lane & 15;
    const int hiA = lane >> 4;

    for (int t = 0; t < T_; ++t) {
        // Prefetch this timestep's gx slice (16 rows x 6144B = 768 cachelines)
        // into cache while the WMMAs below run.
        {
#pragma unroll
            for (int q = 0; q < 3; ++q) {
                const int l = threadIdx.x + q * 256;      // 0..767
                const int m = l / 48, offw = (l % 48) * 32;
                __builtin_prefetch(gx + ((size_t)(b0 + m) * T_ + t) * G3H + offw, 0, 0);
            }
        }

        // ---- gate GEMM: gh = hB * Whh^T + bhh ----
        // 12 N-tiles per wave, processed as 3 groups of 4 with register
        // double-buffering of B fragments (loads for kb+1 issued before the
        // WMMAs of kb -> non-zero loadcnt waits, 4-way WMMA ILP).
#pragma unroll
        for (int g = 0; g < 3; ++g) {
            const int tile0 = (wave * 12 + g * 4) * 16;
            v8f acc[4];
#pragma unroll
            for (int j = 0; j < 4; ++j) acc[j] = (v8f){0.f,0.f,0.f,0.f,0.f,0.f,0.f,0.f};

            v16bf bc[4];
#pragma unroll
            for (int j = 0; j < 4; ++j) bc[j] = load_b_w(Whh, H_, tile0 + j * 16, 0, lane);

#pragma unroll
            for (int kb = 0; kb < 16; ++kb) {
                // A fragment for this kb from LDS h-state (issue ds loads first)
                const bf16_t* p = hB + mA * HB_LD + kb * 32 + hiA * 16;
                v8bf lo = *(const v8bf*)p;
                v8bf hh = *(const v8bf*)(p + 8);
                v16bf bn[4];
                if (kb < 15) {
#pragma unroll
                    for (int j = 0; j < 4; ++j)
                        bn[j] = load_b_w(Whh, H_, tile0 + j * 16, (kb + 1) * 32, lane);
                }
                v16bf a;
#pragma unroll
                for (int i = 0; i < 8; ++i) { a[i] = lo[i]; a[8 + i] = hh[i]; }
#pragma unroll
                for (int j = 0; j < 4; ++j)
                    acc[j] = __builtin_amdgcn_wmma_f32_16x16x32_bf16(
                        false, a, false, bc[j], (short)0, acc[j], false, false);
                if (kb < 15) {
#pragma unroll
                    for (int j = 0; j < 4; ++j) bc[j] = bn[j];
                }
            }

#pragma unroll
            for (int j = 0; j < 4; ++j) {
                const int n = tile0 + j * 16 + (lane & 15);
                const float bv = bhh[n];
#pragma unroll
                for (int v = 0; v < 8; ++v) gh[(v + 8 * hiA) * GH_LD + n] = acc[j][v] + bv;
            }
        }
        __syncthreads();

        // ---- fused elementwise GRU update ----
        for (int u = threadIdx.x; u < 16 * H_; u += blockDim.x) {
            const int m = u >> 9, j = u & (H_ - 1);
            const size_t row = (size_t)(b0 + m) * T_ + t;
            const float* gxp = gx + row * G3H;
            const float r  = sig_f(gxp[j]         + gh[m * GH_LD + j]);
            const float z  = sig_f(gxp[H_ + j]    + gh[m * GH_LD + H_ + j]);
            const float nn = tanh_f(gxp[2*H_ + j] + r * gh[m * GH_LD + 2*H_ + j]);
            const float hp = hF[m * HF_LD + j];
            const float hn = (1.f - z) * nn + z * hp;
            hF[m * HF_LD + j] = hn;
            hB[m * HB_LD + j] = (bf16_t)hn;
            hout[row * H_ + j] = (bf16_t)hn;
        }
        __syncthreads();
    }
}

// ---------------------------------------------------------------------------
// Final FC: out[b] = h2[b, T-1, :] . Wfc + bfc   (O = 1)
// ---------------------------------------------------------------------------
__global__ void fc_kernel(const bf16_t* __restrict__ h2, const float* __restrict__ Wfc,
                          const float* __restrict__ bfc, float* __restrict__ out) {
    const int b = blockIdx.x * blockDim.x + threadIdx.x;
    if (b < B_) {
        const bf16_t* hp = h2 + ((size_t)b * T_ + (T_ - 1)) * H_;
        float s = bfc[0];
        for (int j = 0; j < H_; ++j) s += (float)hp[j] * Wfc[j];
        out[b] = s;
    }
}

// ---------------------------------------------------------------------------
// Launch
// ---------------------------------------------------------------------------
extern "C" void kernel_launch(void* const* d_in, const int* in_sizes, int n_in,
                              void* d_out, int out_size, void* d_ws, size_t ws_size,
                              hipStream_t stream) {
    const float* X     = (const float*)d_in[0];
    const float* Wih0  = (const float*)d_in[1];
    const float* Whh0  = (const float*)d_in[2];
    const float* bih0  = (const float*)d_in[3];
    const float* bhh0  = (const float*)d_in[4];
    const float* Wih1  = (const float*)d_in[5];
    const float* Whh1  = (const float*)d_in[6];
    const float* bih1  = (const float*)d_in[7];
    const float* bhh1  = (const float*)d_in[8];
    const float* Wfc   = (const float*)d_in[9];
    const float* bfc   = (const float*)d_in[10];
    float* out = (float*)d_out;

    const size_t M = (size_t)B_ * T_;           // 32768
    char* ws = (char*)d_ws;
    size_t off = 0;
    float*  gx      = (float*)(ws + off);  off += M * G3H * 4;        // 201.3 MB (shared by both layers)
    bf16_t* Xbf     = (bf16_t*)(ws + off); off += M * D_ * 2;         //   8.4 MB
    bf16_t* h1bf    = (bf16_t*)(ws + off); off += M * H_ * 2;         //  33.6 MB
    bf16_t* h2bf    = (bf16_t*)(ws + off); off += M * H_ * 2;         //  33.6 MB
    bf16_t* Wih0bf  = (bf16_t*)(ws + off); off += (size_t)G3H * D_ * 2;
    bf16_t* Whh0bf  = (bf16_t*)(ws + off); off += (size_t)G3H * H_ * 2;
    bf16_t* Wih1bf  = (bf16_t*)(ws + off); off += (size_t)G3H * H_ * 2;
    bf16_t* Whh1bf  = (bf16_t*)(ws + off); off += (size_t)G3H * H_ * 2;

    // bf16 conversions
    cvt_f32_bf16<<<2048, 256, 0, stream>>>(X,    Xbf,    (int)(M * D_));
    cvt_f32_bf16<<<768,  256, 0, stream>>>(Wih0, Wih0bf, G3H * D_);
    cvt_f32_bf16<<<1536, 256, 0, stream>>>(Whh0, Whh0bf, G3H * H_);
    cvt_f32_bf16<<<1536, 256, 0, stream>>>(Wih1, Wih1bf, G3H * H_);
    cvt_f32_bf16<<<1536, 256, 0, stream>>>(Whh1, Whh1bf, G3H * H_);

    // blocks = (M/128) * (N/64) = 256 * 24 = 6144
    const int gemm_blocks = (int)((M / 128) * (G3H / 64));

    // Layer 0
    wmma_gemm_bias<<<gemm_blocks, 256, 0, stream>>>(Xbf, Wih0bf, bih0, gx, (int)M, G3H, D_);
    gru_scan<<<B_ / 16, 256, SCAN_LDS_BYTES, stream>>>(gx, Whh0bf, bhh0, h1bf);

    // Layer 1
    wmma_gemm_bias<<<gemm_blocks, 256, 0, stream>>>(h1bf, Wih1bf, bih1, gx, (int)M, G3H, H_);
    gru_scan<<<B_ / 16, 256, SCAN_LDS_BYTES, stream>>>(gx, Whh1bf, bhh1, h2bf);

    // FC head
    fc_kernel<<<1, 64, 0, stream>>>(h2bf, Wfc, bfc, out);
}
